// MolecularGraphNetwork_29094108463229
// MI455X (gfx1250) — matrix operand order repaired and checked by reading, not compiled, not run on hardware
//
#include <hip/hip_runtime.h>

#define N_NODES 100000
#define N_EDGES 1600000
#define N_GRAPHS 512

typedef __attribute__((ext_vector_type(16))) __bf16 v16bf;
typedef __attribute__((ext_vector_type(8)))  float  v8f;

#define A_PAD 40   // LDS row stride in bf16 elements (80B = 20 banks, conflict-free)

__device__ __forceinline__ __bf16 f2bf(float f) {
    unsigned int u = __float_as_uint(f);
    u += 0x7FFFu + ((u >> 16) & 1u);           // round-to-nearest-even
    unsigned short h = (unsigned short)(u >> 16);
    return __builtin_bit_cast(__bf16, h);
}

// ---- pack W [K,OUT] f32 -> bf16 in WMMA B-operand layout ----
// P[((kb*(OUT/16) + nt)*32 + lane)*16 + e] = W[kb*32 + 16*(lane/16) + e][nt*16 + lane%16]
__global__ void pack_b_kernel(const float* __restrict__ W, __bf16* __restrict__ P,
                              int K, int OUT) {
    int i = blockIdx.x * blockDim.x + threadIdx.x;
    if (i >= K * OUT) return;
    int e    = i & 15;
    int lane = (i >> 4) & 31;
    int rest = i >> 9;
    int ntiles = OUT >> 4;
    int kb = rest / ntiles;
    int nt = rest - kb * ntiles;
    int k = kb * 32 + ((lane >> 4) << 4) + e;
    int n = (nt << 4) + (lane & 15);
    P[i] = f2bf(W[(size_t)k * OUT + n]);
}

// ---------------- edge scatter-add: agg[dst] += x[src] ----------------
__global__ void edge_scatter_kernel(const float* __restrict__ x, const int* __restrict__ src,
                                    const int* __restrict__ dst, float* __restrict__ agg,
                                    int F, int f4shift, int nwork) {
    int i = blockIdx.x * blockDim.x + threadIdx.x;
    if (i >= nwork) return;
    int e = i >> f4shift;
    int c = i & ((1 << f4shift) - 1);
    int s = src[e], d = dst[e];
    const float4 v = *(const float4*)(x + (size_t)s * F + (size_t)c * 4);
    float* p = agg + (size_t)d * F + (size_t)c * 4;
    atomicAdd(p + 0, v.x); atomicAdd(p + 1, v.y);
    atomicAdd(p + 2, v.z); atomicAdd(p + 3, v.w);
}

// ---------------- global add pool: g[batch[n]] += h[n] ----------------
__global__ void pool_kernel(const float* __restrict__ h, const int* __restrict__ batch,
                            float* __restrict__ g, int nwork) {
    int i = blockIdx.x * blockDim.x + threadIdx.x;
    if (i >= nwork) return;
    int n = i >> 6;
    int c = i & 63;
    int b = batch[n];
    const float4 v = *(const float4*)(h + (size_t)n * 256 + (size_t)c * 4);
    float* p = g + (size_t)b * 256 + (size_t)c * 4;
    atomicAdd(p + 0, v.x); atomicAdd(p + 1, v.y);
    atomicAdd(p + 2, v.z); atomicAdd(p + 3, v.w);
}

// ---------------- per-column sum / sumsq ----------------
__global__ void col_stats_kernel(const float* __restrict__ h, float* __restrict__ sums,
                                 int nrows, int rows_per_block) {
    int c = threadIdx.x;                 // 256 threads = 256 columns
    int r0 = blockIdx.x * rows_per_block;
    int r1 = r0 + rows_per_block; if (r1 > nrows) r1 = nrows;
    float s = 0.f, s2 = 0.f;
    for (int r = r0; r < r1; ++r) {
        float v = h[(size_t)r * 256 + c];
        s += v; s2 += v * v;
    }
    atomicAdd(&sums[c], s);
    atomicAdd(&sums[256 + c], s2);
}

// ---------------- BN apply (biased var, as torch BN fwd) ----------------
__global__ void bn_apply_kernel(const float* __restrict__ h, const float* __restrict__ sums,
                                const float* __restrict__ gamma, const float* __restrict__ beta,
                                float* __restrict__ out, int n) {
    int i = blockIdx.x * blockDim.x + threadIdx.x;
    if (i >= n) return;
    int c = i & 255;
    const float invN = 1.0f / (float)N_NODES;
    float mean = sums[c] * invN;
    float var  = sums[256 + c] * invN - mean * mean;
    out[i] = (h[i] - mean) * rsqrtf(var + 1e-5f) * gamma[c] + beta[c];
}

// ---------------- WMMA bf16 GEMM: C = relu(A @ Bp + bias) ----------------
// A: [M,KDIM] f32 row-major; Bp: packed B layout; C: [M,OUTDIM] f32
// block = 256 threads (8 waves = 2 row-groups x 4 col-groups).
// Block tile: 32 rows x (4*NT*16) cols.  grid = (M/32, OUTDIM/(64*NT))
// A tile double-buffered in LDS; staging load issued at loop top, consumed
// (cvt + ds_store) after the WMMA chain so its latency hides behind compute.
template<int NT, int KDIM, int OUTDIM>
__global__ void __launch_bounds__(256) gemm_bf16_wmma_kernel(
    const float* __restrict__ A, const __bf16* __restrict__ Bp,
    const float* __restrict__ bias, float* __restrict__ C, int M)
{
    __shared__ __bf16 a_lds[2][2 * 16 * A_PAD];   // [buffer][two 16x32 A tiles]
    const int t     = threadIdx.x;
    const int wave  = t >> 5;
    const int waveM = wave >> 2;               // 0..1: row-group
    const int waveN = wave & 3;                // 0..3: col-group
    const int lane  = t & 31;
    const int half  = lane >> 4;
    const int l16   = lane & 15;
    const int rowBase = blockIdx.x * 32 + waveM * 16;
    const int ntilesTotal = OUTDIM >> 4;
    const int ntile0 = (blockIdx.y * 4 + waveN) * NT;  // first 16-col tile of this wave

    // per-thread staging coordinates: 32 rows x 32 cols, 4 f32 per thread
    const int sr  = (t * 4) >> 5;              // 0..31
    const int scc = (t * 4) & 31;
    const float* aThread = A + (size_t)(blockIdx.x * 32 + sr) * KDIM + scc;
    const int ldsOff = ((sr >> 4) * 16 + (sr & 15)) * A_PAD + scc;

    v8f acc[NT];
#pragma unroll
    for (int j = 0; j < NT; ++j) acc[j] = (v8f){};

    constexpr int nkb = KDIM >> 5;

    // prologue: stage tile 0 into buffer 0
    {
        const float4 v = *(const float4*)aThread;
        __bf16* d = &a_lds[0][ldsOff];
        d[0] = f2bf(v.x); d[1] = f2bf(v.y); d[2] = f2bf(v.z); d[3] = f2bf(v.w);
        __builtin_prefetch(aThread + 32, 0, 3);
    }
    __syncthreads();

    for (int kb = 0; kb < nkb; ++kb) {
        const int cur = kb & 1;
        const bool hasNext = (kb + 1 < nkb);

        // issue next A tile's global load FIRST (consumed after the WMMA chain)
        float4 vnext = {};
        if (hasNext) vnext = *(const float4*)(aThread + ((kb + 1) << 5));

        // A operand: lane -> row l16; e -> K = 16*half + e (matches packed B)
        v16bf av;
        const __bf16* arow = &a_lds[cur][(waveM * 16 + l16) * A_PAD];
#pragma unroll
        for (int e = 0; e < 16; ++e) {
            int kk = (half << 4) + e;
            av[e] = arow[kk];
        }
        // B tiles: contiguous 32B per lane (coalesced), then WMMA chain
#pragma unroll
        for (int j = 0; j < NT; ++j) {
            const __bf16* bptr =
                Bp + ((((size_t)kb * ntilesTotal + ntile0 + j) << 5) + lane) * 16;
            v16bf bv = *(const v16bf*)bptr;
            acc[j] = __builtin_amdgcn_wmma_f32_16x16x32_bf16(
                false, av, false, bv, (short)0, acc[j], false, false);
        }

        // convert + store next tile into the other buffer, behind the WMMAs
        if (hasNext) {
            __bf16* d = &a_lds[cur ^ 1][ldsOff];
            d[0] = f2bf(vnext.x); d[1] = f2bf(vnext.y);
            d[2] = f2bf(vnext.z); d[3] = f2bf(vnext.w);
            __builtin_prefetch(aThread + ((kb + 2) << 5), 0, 3);
        }
        __syncthreads();
    }

    // epilogue: bias + relu; C/D layout: VGPR r -> row r + 8*half, col l16
#pragma unroll
    for (int j = 0; j < NT; ++j) {
        int col = ((ntile0 + j) << 4) + l16;
        float bc = bias[col];
#pragma unroll
        for (int r = 0; r < 8; ++r) {
            float v = fmaxf(acc[j][r] + bc, 0.f);
            C[(size_t)(rowBase + r + (half << 3)) * OUTDIM + col] = v;
        }
    }
}

// ---------------- final 256 -> 1 projection ----------------
__global__ void final_kernel(const float* __restrict__ g3, const float* __restrict__ w,
                             const float* __restrict__ b, float* __restrict__ out) {
    int g = blockIdx.x * blockDim.x + threadIdx.x;
    if (g >= N_GRAPHS) return;
    float s = b[0];
    for (int k = 0; k < 256; ++k) s += g3[(size_t)g * 256 + k] * w[k];
    out[g] = s;
}

extern "C" void kernel_launch(void* const* d_in, const int* in_sizes, int n_in,
                              void* d_out, int out_size, void* d_ws, size_t ws_size,
                              hipStream_t stream) {
    const float* d_x       = (const float*)d_in[0];
    const int*   d_edge    = (const int*)d_in[1];
    const int*   d_batch   = (const int*)d_in[2];
    const float* gin1_w1   = (const float*)d_in[3];
    const float* gin1_b1   = (const float*)d_in[4];
    const float* gin1_w2   = (const float*)d_in[5];
    const float* gin1_b2   = (const float*)d_in[6];
    const float* gin_w1    = (const float*)d_in[7];
    const float* gin_b1    = (const float*)d_in[8];
    const float* gin_w2    = (const float*)d_in[9];
    const float* gin_b2    = (const float*)d_in[10];
    const float* bn_gamma  = (const float*)d_in[11];
    const float* bn_beta   = (const float*)d_in[12];
    const float* fc_xd_w   = (const float*)d_in[13];
    const float* fc_xd_b   = (const float*)d_in[14];
    const float* fc1_w     = (const float*)d_in[15];
    const float* fc1_b     = (const float*)d_in[16];
    const float* fc2_w     = (const float*)d_in[17];
    const float* fc2_b     = (const float*)d_in[18];
    const float* out_w     = (const float*)d_in[19];
    const float* out_b     = (const float*)d_in[20];

    const int* e_src = d_edge;
    const int* e_dst = d_edge + N_EDGES;

    // ---- workspace carve-up (all region sizes are multiples of 256B) ----
    char* p = (char*)d_ws;
    const size_t n256 = (size_t)N_NODES * 256;
    float* xcur = (float*)p; p += n256 * 4;
    float* agg  = (float*)p; p += n256 * 4;
    float* t1   = (float*)p; p += n256 * 4;
    float* sums = (float*)p; p += 2048;
    float* g    = (float*)p; p += (size_t)N_GRAPHS * 256 * 4;
    float* g1   = (float*)p; p += (size_t)N_GRAPHS * 128 * 4;
    float* g2   = (float*)p; p += (size_t)N_GRAPHS * 1024 * 4;
    float* g3   = (float*)p; p += (size_t)N_GRAPHS * 256 * 4;
    __bf16* w_gin1_1 = (__bf16*)p; p += (size_t)128 * 256 * 2;
    __bf16* w_gin1_2 = (__bf16*)p; p += (size_t)256 * 256 * 2;
    __bf16* w_gin_1  = (__bf16*)p; p += (size_t)5 * 256 * 256 * 2;
    __bf16* w_gin_2  = (__bf16*)p; p += (size_t)5 * 256 * 256 * 2;
    __bf16* w_fcxd   = (__bf16*)p; p += (size_t)256 * 128 * 2;
    __bf16* w_fc1    = (__bf16*)p; p += (size_t)128 * 1024 * 2;
    __bf16* w_fc2    = (__bf16*)p; p += (size_t)1024 * 256 * 2;
    (void)ws_size; (void)n_in; (void)in_sizes; (void)out_size;

    // ---- pack weights to WMMA bf16 B layout (every call; deterministic) ----
    auto pack = [&](const float* s, __bf16* d, int K, int OUT) {
        int n = K * OUT;
        pack_b_kernel<<<(n + 255) / 256, 256, 0, stream>>>(s, d, K, OUT);
    };
    pack(gin1_w1, w_gin1_1, 128, 256);
    pack(gin1_w2, w_gin1_2, 256, 256);
    for (int i = 0; i < 5; ++i) {
        pack(gin_w1 + (size_t)i * 256 * 256, w_gin_1 + (size_t)i * 256 * 256, 256, 256);
        pack(gin_w2 + (size_t)i * 256 * 256, w_gin_2 + (size_t)i * 256 * 256, 256, 256);
    }
    pack(fc_xd_w, w_fcxd, 256, 128);
    pack(fc1_w,   w_fc1,  128, 1024);
    pack(fc2_w,   w_fc2,  1024, 256);

    // ---- 6 GIN layers ----
    for (int i = 0; i < 6; ++i) {
        const float* X = (i == 0) ? d_x : xcur;
        const int F = (i == 0) ? 128 : 256;
        const int f4shift = (i == 0) ? 5 : 6;          // log2(F/4)

        // agg = X (self term), then agg[dst] += X[src]
        hipMemcpyAsync(agg, X, (size_t)N_NODES * F * 4, hipMemcpyDeviceToDevice, stream);
        int nwork = N_EDGES * (F / 4);
        edge_scatter_kernel<<<(nwork + 255) / 256, 256, 0, stream>>>(
            X, e_src, e_dst, agg, F, f4shift, nwork);

        const __bf16* W1 = (i == 0) ? w_gin1_1 : (w_gin_1 + (size_t)(i - 1) * 256 * 256);
        const __bf16* W2 = (i == 0) ? w_gin1_2 : (w_gin_2 + (size_t)(i - 1) * 256 * 256);
        const float*  B1 = (i == 0) ? gin1_b1  : (gin_b1 + (size_t)(i - 1) * 256);
        const float*  B2 = (i == 0) ? gin1_b2  : (gin_b2 + (size_t)(i - 1) * 256);

        // t1 = relu(agg @ W1 + B1); agg = relu(t1 @ W2 + B2)
        if (i == 0)
            gemm_bf16_wmma_kernel<4, 128, 256><<<dim3(N_NODES / 32, 1), 256, 0, stream>>>(
                agg, W1, B1, t1, N_NODES);
        else
            gemm_bf16_wmma_kernel<4, 256, 256><<<dim3(N_NODES / 32, 1), 256, 0, stream>>>(
                agg, W1, B1, t1, N_NODES);
        gemm_bf16_wmma_kernel<4, 256, 256><<<dim3(N_NODES / 32, 1), 256, 0, stream>>>(
            t1, W2, B2, agg, N_NODES);

        // BatchNorm (batch stats) -> xcur
        hipMemsetAsync(sums, 0, 2048, stream);
        col_stats_kernel<<<500, 256, 0, stream>>>(agg, sums, N_NODES, 200);
        bn_apply_kernel<<<((int)n256 + 255) / 256, 256, 0, stream>>>(
            agg, sums, bn_gamma + (size_t)i * 256, bn_beta + (size_t)i * 256,
            xcur, (int)n256);
    }

    // ---- global add pool ----
    hipMemsetAsync(g, 0, (size_t)N_GRAPHS * 256 * 4, stream);
    int pwork = N_NODES * 64;
    pool_kernel<<<(pwork + 255) / 256, 256, 0, stream>>>(xcur, d_batch, g, pwork);

    // ---- head ----
    gemm_bf16_wmma_kernel<2, 256, 128><<<dim3(N_GRAPHS / 32, 1), 256, 0, stream>>>(
        g, w_fcxd, fc_xd_b, g1, N_GRAPHS);
    gemm_bf16_wmma_kernel<4, 128, 1024><<<dim3(N_GRAPHS / 32, 4), 256, 0, stream>>>(
        g1, w_fc1, fc1_b, g2, N_GRAPHS);
    gemm_bf16_wmma_kernel<4, 1024, 256><<<dim3(N_GRAPHS / 32, 1), 256, 0, stream>>>(
        g2, w_fc2, fc2_b, g3, N_GRAPHS);
    final_kernel<<<2, 256, 0, stream>>>(g3, out_w, out_b, (float*)d_out);
}
